// RhythmALIF_1503238554049
// MI455X (gfx1250) — matrix operand order repaired and checked by reading, not compiled, not run on hardware
//
#include <hip/hip_runtime.h>

// ---------------------------------------------------------------------------
// RhythmALIF scan for MI455X (gfx1250, wave32, WMMA).
//   B=512, I=128, H=256, O=64, T=512.
// Decomposition: 32 workgroups x 16 batch rows; 16 waves/WG, each wave owns a
// 16-column slice of H. Entire T=512 scan runs inside one kernel; spikes are
// exchanged between waves through double-buffered f16 LDS tiles; weights are
// f16 in global memory (L2-resident, 608KB total); x is pre-transposed to
// f16 [T][B][I] so input-projection A-fragments are contiguous.
//
// v2: weight base pointers are laundered through an empty volatile asm inside
// the t-loop so the B-fragment loads stay IN the loop (L2/WGP$ hits) instead
// of being hoisted into 288 VGPRs and spilled to scratch (seen in v1 disasm
// as "scratch_load_b128 ... Folded Reload"). Also prefetch next x tile.
// ---------------------------------------------------------------------------

typedef __attribute__((ext_vector_type(16))) _Float16 v16h;
typedef __attribute__((ext_vector_type(8)))  _Float16 v8h;
typedef __attribute__((ext_vector_type(8)))  float    v8f;

#define BB 512
#define II 128
#define HH 256
#define OO 64
#define TT 512
#define HP 264          // padded LDS row stride in halves (16B aligned, 4-bank skew)

// workspace layout (in _Float16 elements)
#define OFF_WI1 0                       // H*I   = 32768
#define OFF_WH1 32768                   // H*H   = 65536
#define OFF_WI2 98304
#define OFF_WH2 163840
#define OFF_WI3 229376
#define OFF_WO  294912                  // O*H   = 16384
#define OFF_XT  311296                  // T*B*I = 33554432

// Launder a pointer through an empty volatile asm: the value is re-defined
// every execution, so loads derived from it cannot be hoisted out of the
// timestep loop (prevents the register-spill catastrophe of v1).
__device__ __forceinline__ const _Float16* opaque(const _Float16* p) {
    asm volatile("" : "+s"(p));
    return p;
}

// --- fragment loaders -------------------------------------------------------
// A: 16(M) x 32(K) f16, row-major source with row stride `ld` (halves).
// lane<16: row=lane,   K = k0..k0+15
// lane>=16: row=lane-16,K = k0+16..k0+31
__device__ __forceinline__ v16h load_a16x32(const _Float16* base, int ld, int k0, int lane) {
    int row = lane & 15;
    int kb  = k0 + ((lane >> 4) << 4);
    const _Float16* p = base + row * ld + kb;
    v8h lo = *(const v8h*)(p);
    v8h hi = *(const v8h*)(p + 8);
    return __builtin_shufflevector(lo, hi, 0,1,2,3,4,5,6,7,8,9,10,11,12,13,14,15);
}

// B: 32(K) x 16(N) f16 where B[k][n] = W[n][k], W row-major [N][ld].
// lane<16: n=n0+lane, K = k0..k0+15 ; lane>=16: K = k0+16..k0+31. Contiguous.
__device__ __forceinline__ v16h load_b32x16(const _Float16* w, int ld, int n0, int k0, int lane) {
    int n  = n0 + (lane & 15);
    int kb = k0 + ((lane >> 4) << 4);
    return *(const v16h*)(w + n * ld + kb);
}

__device__ __forceinline__ v8f wmma_f16(v16h a, v16h b, v8f c) {
    return __builtin_amdgcn_wmma_f32_16x16x32_f16(false, a, false, b, (short)0, c, false, false);
}

// ALIF cell update on the 16x16 C layout: each lane holds 8 rows of one column.
__device__ __forceinline__ void alif_cell(const v8f& h, float bias,
                                          float* mem, float* spk, float* bad,
                                          float alpha, float ro, float mt) {
#pragma unroll
    for (int i = 0; i < 8; ++i) {
        float b   = ro * bad[i] + (1.0f - ro) * spk[i];
        bad[i]    = b;
        float Bth = 0.01f + 1.8f * b;
        float inp = h[i] + bias;
        float mn  = mem[i] * alpha + inp - Bth * spk[i];
        float m   = (mt == 0.0f) ? mem[i] : mn;
        mem[i]    = m;
        spk[i]    = ((m - Bth) > 0.0f ? 1.0f : 0.0f) * mt;
    }
}

// --- stage 1: weight conversion f32 -> f16 ---------------------------------
__global__ void convert_weights(const float* __restrict__ wi1, const float* __restrict__ wh1,
                                const float* __restrict__ wi2, const float* __restrict__ wh2,
                                const float* __restrict__ wi3, const float* __restrict__ wo,
                                _Float16* __restrict__ dst) {
    int idx = blockIdx.x * blockDim.x + threadIdx.x;
    float v;
    if      (idx < 32768)  v = wi1[idx];
    else if (idx < 98304)  v = wh1[idx - 32768];
    else if (idx < 163840) v = wi2[idx - 98304];
    else if (idx < 229376) v = wh2[idx - 163840];
    else if (idx < 294912) v = wi3[idx - 229376];
    else if (idx < 311296) v = wo [idx - 294912];
    else return;
    dst[idx] = (_Float16)v;
}

// --- stage 2: x [B][I][T] f32 -> xT [T][B][I] f16 (LDS-tiled transpose) ----
__global__ void transpose_x(const float* __restrict__ x, _Float16* __restrict__ xT) {
    __shared__ _Float16 tile[16][68];
    int b  = blockIdx.z;
    int i0 = blockIdx.y * 16;
    int t0 = blockIdx.x * 64;
    int tid = threadIdx.x; // 256
#pragma unroll
    for (int c = 0; c < 4; ++c) {
        int ii = c * 4 + (tid >> 6);
        int tt = tid & 63;
        tile[ii][tt] = (_Float16)x[((size_t)b * II + i0 + ii) * TT + t0 + tt];
    }
    __syncthreads();
#pragma unroll
    for (int c = 0; c < 4; ++c) {
        int tt = c * 16 + (tid >> 4);
        int ii = tid & 15;
        xT[((size_t)(t0 + tt) * BB + b) * II + i0 + ii] = tile[ii][tt];
    }
}

// --- main persistent scan kernel -------------------------------------------
__global__ __launch_bounds__(512, 1)
void rhythm_alif_scan(const _Float16* __restrict__ xT,
                      const _Float16* __restrict__ wi1_, const _Float16* __restrict__ wh1_,
                      const _Float16* __restrict__ wi2_, const _Float16* __restrict__ wh2_,
                      const _Float16* __restrict__ wi3_, const _Float16* __restrict__ wo,
                      const float* __restrict__ h1i, const float* __restrict__ h2i,
                      const float* __restrict__ h3i,
                      const float* __restrict__ bi1, const float* __restrict__ bh1,
                      const float* __restrict__ bi2, const float* __restrict__ bh2,
                      const float* __restrict__ bi3, const float* __restrict__ bo,
                      const float* __restrict__ ta1, const float* __restrict__ ta2,
                      const float* __restrict__ ta3,
                      const float* __restrict__ tm1, const float* __restrict__ tm2,
                      const float* __restrict__ tm3,
                      const float* __restrict__ mask1, const float* __restrict__ mask2,
                      const float* __restrict__ mask3,
                      float* __restrict__ out) {
    __shared__ alignas(16) _Float16 s1buf[2][16 * HP];
    __shared__ alignas(16) _Float16 s2buf[2][16 * HP];
    __shared__ alignas(16) _Float16 s3buf[16 * HP];

    const int tid  = threadIdx.x;
    const int wv   = tid >> 5;          // 16 waves
    const int lane = tid & 31;
    const int n0   = wv * 16;           // this wave's H-column slice
    const int hcol = n0 + (lane & 15);  // this lane's column
    const int hi16 = lane >> 4;         // row-half selector in C layout
    const int bb   = blockIdx.x * 16;   // batch tile base

    // per-column constants
    const float alpha1 = expf(-1.0f / tm1[hcol]);
    const float alpha2 = expf(-1.0f / tm2[hcol]);
    const float alpha3 = expf(-1.0f / tm3[hcol]);
    const float ro1    = expf(-1.0f / ta1[hcol]);
    const float ro2    = expf(-1.0f / ta2[hcol]);
    const float ro3    = expf(-1.0f / ta3[hcol]);
    const float bias1  = bi1[hcol] + bh1[hcol];
    const float bias2  = bi2[hcol] + bh2[hcol];
    const float bias3  = bi3[hcol];
    const float* m1p = mask1 + (size_t)hcol * TT;
    const float* m2p = mask2 + (size_t)hcol * TT;
    const float* m3p = mask3 + (size_t)hcol * TT;

    // per-lane state: 8 rows x this lane's column for each layer
    float mem1[8], spk1[8], bad1[8];
    float mem2[8], spk2[8], bad2[8];
    float mem3[8], spk3[8], bad3[8];
#pragma unroll
    for (int r = 0; r < 8; ++r) {
        int row = r + 8 * hi16;
        int g   = (bb + row) * HH + hcol;
        mem1[r] = h1i[g]; spk1[r] = mem1[r]; bad1[r] = 0.01f;
        mem2[r] = h2i[g]; spk2[r] = mem2[r]; bad2[r] = 0.01f;
        mem3[r] = h3i[g]; spk3[r] = mem3[r]; bad3[r] = 0.01f;
        s1buf[0][row * HP + hcol] = (_Float16)spk1[r];
        s2buf[0][row * HP + hcol] = (_Float16)spk2[r];
    }
    __syncthreads();

    for (int t = 0; t < TT; ++t) {
        const int cur = t & 1, nxt = cur ^ 1;

        // Re-launder weight pointers every iteration: keeps the B-fragment
        // loads inside the loop (they hit WGP$/L2; weights are only 608KB)
        // instead of being hoisted+spilled to scratch.
        const _Float16* wi1 = opaque(wi1_);
        const _Float16* wh1 = opaque(wh1_);
        const _Float16* wi2 = opaque(wi2_);
        const _Float16* wh2 = opaque(wh2_);
        const _Float16* wi3 = opaque(wi3_);

        // ---- phase A: h1 = x_t @ Wi1^T + s1 @ Wh1^T (+bias), cell1 --------
        v8f c = {};
        {
            const _Float16* xrow = xT + (size_t)(t * BB + bb) * II;
#pragma unroll
            for (int k0 = 0; k0 < II; k0 += 32) {
                v16h a = load_a16x32(xrow, II, k0, lane);
                v16h b = load_b32x16(wi1, II, n0, k0, lane);
                c = wmma_f16(a, b, c);
            }
#pragma unroll
            for (int k0 = 0; k0 < HH; k0 += 32) {
                v16h a = load_a16x32(&s1buf[cur][0], HP, k0, lane);
                v16h b = load_b32x16(wh1, HH, n0, k0, lane);
                c = wmma_f16(a, b, c);
            }
            // prefetch next timestep's x tile toward this WGP (HBM stream)
            if (t + 1 < TT && tid < 256) {
                const _Float16* xnext = xT + (size_t)((t + 1) * BB + bb) * II;
                __builtin_prefetch(xnext + tid * 8, 0, 3);
            }
        }
        alif_cell(c, bias1, mem1, spk1, bad1, alpha1, ro1, m1p[t]);
#pragma unroll
        for (int r = 0; r < 8; ++r)
            s1buf[nxt][(r + 8 * hi16) * HP + hcol] = (_Float16)spk1[r];
        __syncthreads();

        // ---- phase B: h2 = s1' @ Wi2^T + s2 @ Wh2^T (+bias), cell2 --------
        c = v8f{};
#pragma unroll
        for (int k0 = 0; k0 < HH; k0 += 32) {
            v16h a = load_a16x32(&s1buf[nxt][0], HP, k0, lane);
            v16h b = load_b32x16(wi2, HH, n0, k0, lane);
            c = wmma_f16(a, b, c);
        }
#pragma unroll
        for (int k0 = 0; k0 < HH; k0 += 32) {
            v16h a = load_a16x32(&s2buf[cur][0], HP, k0, lane);
            v16h b = load_b32x16(wh2, HH, n0, k0, lane);
            c = wmma_f16(a, b, c);
        }
        alif_cell(c, bias2, mem2, spk2, bad2, alpha2, ro2, m2p[t]);
#pragma unroll
        for (int r = 0; r < 8; ++r)
            s2buf[nxt][(r + 8 * hi16) * HP + hcol] = (_Float16)spk2[r];
        __syncthreads();

        // ---- phase C: h3 = s2' @ Wi3^T (+bias), cell3 ---------------------
        c = v8f{};
#pragma unroll
        for (int k0 = 0; k0 < HH; k0 += 32) {
            v16h a = load_a16x32(&s2buf[nxt][0], HP, k0, lane);
            v16h b = load_b32x16(wi3, HH, n0, k0, lane);
            c = wmma_f16(a, b, c);
        }
        alif_cell(c, bias3, mem3, spk3, bad3, alpha3, ro3, m3p[t]);
        if (t == TT - 1) {
#pragma unroll
            for (int r = 0; r < 8; ++r)
                s3buf[(r + 8 * hi16) * HP + hcol] = (_Float16)spk3[r];
        }
    }
    __syncthreads();

    // ---- output: sigmoid(s3 @ Wo^T + bo), only last timestep, O=64 --------
    if (wv < 4) {
        v8f c = {};
#pragma unroll
        for (int k0 = 0; k0 < HH; k0 += 32) {
            v16h a = load_a16x32(&s3buf[0], HP, k0, lane);
            v16h b = load_b32x16(wo, HH, n0, k0, lane);
            c = wmma_f16(a, b, c);
        }
        const int ocol = n0 + (lane & 15);
        const float bia = bo[ocol];
#pragma unroll
        for (int r = 0; r < 8; ++r) {
            int row = r + 8 * hi16;
            float v = c[r] + bia;
            out[(size_t)(bb + row) * OO + ocol] = 1.0f / (1.0f + expf(-v));
        }
    }
}

// ---------------------------------------------------------------------------
extern "C" void kernel_launch(void* const* d_in, const int* in_sizes, int n_in,
                              void* d_out, int out_size, void* d_ws, size_t ws_size,
                              hipStream_t stream) {
    (void)in_sizes; (void)n_in; (void)out_size;
    const float* x    = (const float*)d_in[0];
    const float* h1i  = (const float*)d_in[1];
    const float* h2i  = (const float*)d_in[2];
    const float* h3i  = (const float*)d_in[3];
    const float* Wi1  = (const float*)d_in[4];
    const float* bi1  = (const float*)d_in[5];
    const float* Wh1  = (const float*)d_in[6];
    const float* bh1  = (const float*)d_in[7];
    const float* Wi2  = (const float*)d_in[8];
    const float* bi2  = (const float*)d_in[9];
    const float* Wh2  = (const float*)d_in[10];
    const float* bh2  = (const float*)d_in[11];
    const float* Wi3  = (const float*)d_in[12];
    const float* bi3  = (const float*)d_in[13];
    const float* Wo   = (const float*)d_in[14];
    const float* bo   = (const float*)d_in[15];
    const float* ta1  = (const float*)d_in[16];
    const float* ta2  = (const float*)d_in[17];
    const float* ta3  = (const float*)d_in[18];
    const float* tm1  = (const float*)d_in[19];
    const float* tm2  = (const float*)d_in[20];
    const float* tm3  = (const float*)d_in[21];
    const float* mk1  = (const float*)d_in[22];
    const float* mk2  = (const float*)d_in[23];
    const float* mk3  = (const float*)d_in[24];

    _Float16* wsH = (_Float16*)d_ws;   // needs ~68MB of scratch
    (void)ws_size;

    // stage 1: f32 -> f16 weight conversion (311296 elements)
    convert_weights<<<1216, 256, 0, stream>>>(Wi1, Wh1, Wi2, Wh2, Wi3, Wo, wsH);
    // stage 2: x [B][I][T] -> xT [T][B][I] f16
    transpose_x<<<dim3(TT / 64, II / 16, BB), 256, 0, stream>>>(x, wsH + OFF_XT);
    // stage 3: full 512-step scan, 32 WGs x 512 threads (16 waves)
    rhythm_alif_scan<<<BB / 16, 512, 0, stream>>>(
        wsH + OFF_XT,
        wsH + OFF_WI1, wsH + OFF_WH1, wsH + OFF_WI2, wsH + OFF_WH2,
        wsH + OFF_WI3, wsH + OFF_WO,
        h1i, h2i, h3i, bi1, bh1, bi2, bh2, bi3, bo,
        ta1, ta2, ta3, tm1, tm2, tm3, mk1, mk2, mk3,
        (float*)d_out);
}
